// SuperPoint_8555574854428
// MI455X (gfx1250) — matrix operand order repaired and checked by reading, not compiled, not run on hardware
//
#include <hip/hip_runtime.h>
#include <hip/hip_bf16.h>

// ---------------- types ----------------
typedef __attribute__((ext_vector_type(16))) _Float16 v16h;
typedef __attribute__((ext_vector_type(8)))  _Float16 v8h;
typedef __attribute__((ext_vector_type(8)))  float    v8f;

union AFrag { v16h v; v8h h[2]; };

#define TM  128
#define TN  64
#define TKC 32

#define SP_MAXK   1024
#define SP_CAP    32768
#define SP_BORDER 4
#define SP_THR    0.005f
#define SP_NMSR   4

// ---------------- implicit-GEMM conv (WMMA f16 -> f32 acc), NHWC ----------------
// Activations: [B][H*W][Cin] f16.  Weights: [Cout][K] f16 with K ordered as
// k = (ky*ksz+kx)*Cin + cin  (so 16 consecutive k share one conv tap -> b128 loads).
__global__ __launch_bounds__(256)
void conv_wmma(const _Float16* __restrict__ in, const _Float16* __restrict__ wgt,
               const float* __restrict__ bias, _Float16* __restrict__ out16,
               float* __restrict__ out32,
               int Cin, int CinShift, int Cout, int H, int W, int ksz, int pad, int relu)
{
    const int HW = H * W;
    const int K  = Cin * ksz * ksz;
    const int b  = blockIdx.z;
    const int m0 = blockIdx.x * TM;
    const int n0 = blockIdx.y * TN;

    __shared__ __align__(16) _Float16 As[2][TM][TKC + 8];
    __shared__ __align__(16) _Float16 Bs[2][TN][TKC + 8];

    const int tid  = threadIdx.x;
    const int lane = tid & 31;
    const int wave = tid >> 5;
    const int wm   = wave & 3;   // 4 waves along M
    const int wn   = wave >> 2;  // 2 waves along N
    const int l16  = lane & 15;
    const int lhi  = lane >> 4;  // 0 or 1

    const _Float16* inb = in + (size_t)b * HW * Cin;

    v8f acc[2][2];
    #pragma unroll
    for (int i = 0; i < 2; ++i)
        #pragma unroll
        for (int j = 0; j < 2; ++j)
            acc[i][j] = {};

    const int ktiles = (K + TKC - 1) / TKC;
    const bool vecA = (CinShift >= 4);      // Cin multiple of 16 (64/128/256)
    const bool vecB = ((K & 7) == 0);

    // A-tile thread mapping: r = tid>>1 (pixel row), kh = (tid&1)*16
    const int ar = tid >> 1;
    const int ak = (tid & 1) * 16;
    const int ap_pix = m0 + ar;
    const int ay = ap_pix / W;
    const int ax = ap_pix - ay * W;
    // B-tile thread mapping: n-row = tid>>2, 8 halves at seg
    const int bn  = tid >> 2;
    const int bsg = (tid & 3) * 8;
    const int bnn = n0 + bn;

    auto stage = [&](int buf, int kt) {
        // ---- A tile: 128 pixels x 32 K (im2col) ----
        if (vecA) {
            int kg0 = kt * TKC + ak;
            int rem = kg0 >> CinShift;          // conv tap index, same for 16 k
            int ci0 = kg0 & (Cin - 1);
            int ky  = rem / ksz;
            int kx  = rem - ky * ksz;
            v8h d0 = {}, d1 = {};
            if (ap_pix < HW) {
                int yy = ay + ky - pad;
                int xx = ax + kx - pad;
                if (yy >= 0 && yy < H && xx >= 0 && xx < W) {
                    const _Float16* src = inb + ((size_t)yy * W + xx) * Cin + ci0;
                    d0 = *(const v8h*)src;
                    d1 = *(const v8h*)(src + 8);
                }
            }
            *(v8h*)&As[buf][ar][ak]     = d0;
            *(v8h*)&As[buf][ar][ak + 8] = d1;
            // distance-2 prefetch of the activation stream (global_prefetch_b8)
            int kgp = kg0 + 2 * TKC;
            if (kgp < K && ap_pix < HW) {
                int remp = kgp >> CinShift;
                int cip  = kgp & (Cin - 1);
                int kyp  = remp / ksz;
                int kxp  = remp - kyp * ksz;
                int yyp  = ay + kyp - pad;
                int xxp  = ax + kxp - pad;
                if (yyp >= 0 && yyp < H && xxp >= 0 && xxp < W)
                    __builtin_prefetch(inb + ((size_t)yyp * W + xxp) * Cin + cip, 0, 1);
            }
        } else {                                 // conv1a only (Cin==1, K==9)
            #pragma unroll
            for (int j = 0; j < 16; ++j) {
                int kg = kt * TKC + ak + j;
                _Float16 val = (_Float16)0.f;
                if (ap_pix < HW && kg < K) {
                    int rem = kg >> CinShift;
                    int ky  = rem / ksz;
                    int kx  = rem - ky * ksz;
                    int yy  = ay + ky - pad;
                    int xx  = ax + kx - pad;
                    if (yy >= 0 && yy < H && xx >= 0 && xx < W)
                        val = inb[((size_t)yy * W + xx) * Cin];
                }
                As[buf][ar][ak + j] = val;
            }
        }
        // ---- B tile: 64 cout x 32 K ----
        if (vecB) {
            int kg0 = kt * TKC + bsg;
            v8h bv = {};
            if (bnn < Cout && kg0 < K)
                bv = *(const v8h*)(wgt + (size_t)bnn * K + kg0);
            *(v8h*)&Bs[buf][bn][bsg] = bv;
            int kgp = kg0 + 2 * TKC;
            if (kgp < K && bnn < Cout)
                __builtin_prefetch(wgt + (size_t)bnn * K + kgp, 0, 1);
        } else {
            #pragma unroll
            for (int j = 0; j < 8; ++j) {
                int kg = kt * TKC + bsg + j;
                _Float16 v = (_Float16)0.f;
                if (bnn < Cout && kg < K) v = wgt[(size_t)bnn * K + kg];
                Bs[buf][bn][bsg + j] = v;
            }
        }
    };

    stage(0, 0);
    __syncthreads();

    for (int kt = 0; kt < ktiles; ++kt) {
        int buf = kt & 1;
        if (kt + 1 < ktiles) stage(buf ^ 1, kt + 1);

        AFrag a[2], bf[2];
        #pragma unroll
        for (int sm = 0; sm < 2; ++sm) {
            int row = wm * 32 + sm * 16 + l16;
            const _Float16* ap = &As[buf][row][0];
            int off0 = lhi ? 8 : 0;                 // ISA A 16x32 f16 layout
            a[sm].h[0] = *(const v8h*)(ap + off0);
            a[sm].h[1] = *(const v8h*)(ap + off0 + 16);
        }
        #pragma unroll
        for (int sn = 0; sn < 2; ++sn) {
            int col = wn * 32 + sn * 16 + l16;
            const _Float16* bp = &Bs[buf][col][0];
            int off = lhi ? 8 : 0;                  // ISA B 32x16 f16 layout
            bf[sn].h[0] = *(const v8h*)(bp + off);
            bf[sn].h[1] = *(const v8h*)(bp + off + 16);
        }
        #pragma unroll
        for (int sm = 0; sm < 2; ++sm)
            #pragma unroll
            for (int sn = 0; sn < 2; ++sn)
                acc[sm][sn] = __builtin_amdgcn_wmma_f32_16x16x32_f16(
                    false, a[sm].v, false, bf[sn].v,
                    (short)0, acc[sm][sn], false, false);
        __syncthreads();
    }

    // epilogue: bias + relu + NHWC store (D layout: N=l16, M rows = lhi*8 + r)
    #pragma unroll
    for (int sm = 0; sm < 2; ++sm) {
        #pragma unroll
        for (int sn = 0; sn < 2; ++sn) {
            int n = n0 + wn * 32 + sn * 16 + l16;
            if (n >= Cout) continue;
            float bv = bias ? bias[n] : 0.f;
            #pragma unroll
            for (int r = 0; r < 8; ++r) {
                int m = m0 + wm * 32 + sm * 16 + lhi * 8 + r;
                if (m >= HW) continue;
                float v = acc[sm][sn][r] + bv;
                if (relu) v = fmaxf(v, 0.f);
                size_t o = ((size_t)b * HW + m) * Cout + n;
                if (out16) out16[o] = (_Float16)v;
                if (out32) out32[o] = v;
            }
        }
    }
}

// ---------------- helpers ----------------
__global__ void f32_to_f16(const float* __restrict__ in, _Float16* __restrict__ out, int n)
{
    int i = blockIdx.x * blockDim.x + threadIdx.x;
    if (i < n) out[i] = (_Float16)in[i];
}

// OIHW f32 -> [O][(ky*ksz+kx)*Cin + i] f16
__global__ void convert_weights(const float* __restrict__ w, _Float16* __restrict__ out,
                                int Cin, int Cout, int ksz)
{
    int K = Cin * ksz * ksz;
    int total = Cout * K;
    int i = blockIdx.x * blockDim.x + threadIdx.x;
    if (i >= total) return;
    int o = i / K;
    int kn = i - o * K;
    int rem = kn / Cin;
    int ci  = kn - rem * Cin;
    int ky  = rem / ksz;
    int kx  = rem - ky * ksz;
    out[i] = (_Float16)w[(((size_t)o * Cin + ci) * ksz + ky) * ksz + kx];
}

__global__ void maxpool2_f16(const _Float16* __restrict__ in, _Float16* __restrict__ out,
                             int C, int cshift, int H, int W)
{
    int Ho = H >> 1, Wo = W >> 1;
    int b = blockIdx.z;
    int i = blockIdx.x * blockDim.x + threadIdx.x;
    if (i >= Ho * Wo * C) return;
    int pix = i >> cshift;
    int c   = i & (C - 1);
    int y = pix / Wo, x = pix - y * Wo;
    const _Float16* s = in + (size_t)b * H * W * C;
    int y2 = y * 2, x2 = x * 2;
    float m = fmaxf(
        fmaxf((float)s[((size_t)y2 * W + x2) * C + c],
              (float)s[((size_t)y2 * W + x2 + 1) * C + c]),
        fmaxf((float)s[((size_t)(y2 + 1) * W + x2) * C + c],
              (float)s[((size_t)(y2 + 1) * W + x2 + 1) * C + c]));
    out[((size_t)b * Ho * Wo + pix) * C + c] = (_Float16)m;
}

// semi is NHWC: [B][P][65]
__global__ void softmax_heat(const float* __restrict__ semi, float* __restrict__ heat,
                             int Hc, int Wc)
{
    int b = blockIdx.z;
    int i = blockIdx.x * blockDim.x + threadIdx.x;
    int P = Hc * Wc;
    if (i >= P) return;
    int h = i / Wc, w = i - h * Wc;
    const float* s = semi + ((size_t)b * P + i) * 65;
    float mx = -1e30f;
    for (int c = 0; c < 65; ++c) mx = fmaxf(mx, s[c]);
    float sum = 0.f;
    for (int c = 0; c < 65; ++c) sum += expf(s[c] - mx);
    float inv = 1.f / sum;
    int Hf = Hc * 8, Wf = Wc * 8;
    float* hb = heat + (size_t)b * Hf * Wf;
    for (int c = 0; c < 64; ++c) {
        float v = expf(s[c] - mx) * inv;
        int cy = c >> 3, cx = c & 7;
        hb[(size_t)(h * 8 + cy) * Wf + (w * 8 + cx)] = v;
    }
}

__global__ void maxfilt_x(const float* __restrict__ in, float* __restrict__ out,
                          int H, int W, int r)
{
    int b = blockIdx.z;
    int i = blockIdx.x * blockDim.x + threadIdx.x;
    if (i >= H * W) return;
    int y = i / W, x = i - y * W;
    const float* p = in + (size_t)b * H * W;
    float m = -1e30f;
    for (int dx = -r; dx <= r; ++dx) {
        int xx = x + dx;
        if (xx >= 0 && xx < W) m = fmaxf(m, p[(size_t)y * W + xx]);
    }
    out[(size_t)b * H * W + i] = m;
}

__global__ void maxfilt_y(const float* __restrict__ in, float* __restrict__ out,
                          int H, int W, int r)
{
    int b = blockIdx.z;
    int i = blockIdx.x * blockDim.x + threadIdx.x;
    if (i >= H * W) return;
    int y = i / W, x = i - y * W;
    const float* p = in + (size_t)b * H * W;
    float m = -1e30f;
    for (int dy = -r; dy <= r; ++dy) {
        int yy = y + dy;
        if (yy >= 0 && yy < H) m = fmaxf(m, p[(size_t)yy * W + x]);
    }
    out[(size_t)b * H * W + i] = m;
}

__global__ void k_eq_mask(const float* a, const float* bpool, float* mask, int n)
{
    int i = blockIdx.x * blockDim.x + threadIdx.x;
    if (i < n) mask[i] = (a[i] == bpool[i]) ? 1.f : 0.f;
}

__global__ void k_suppscore(const float* pm, const float* heat, float* ss, int n)
{
    int i = blockIdx.x * blockDim.x + threadIdx.x;
    if (i < n) ss[i] = (pm[i] > 0.f) ? 0.f : heat[i];
}

__global__ void k_update_mask(float* maskf, const float* ss, const float* ps,
                              const float* pm, int n)
{
    int i = blockIdx.x * blockDim.x + threadIdx.x;
    if (i >= n) return;
    bool supp   = pm[i] > 0.f;
    bool newmax = (ss[i] == ps[i]) && !supp;
    maskf[i] = (maskf[i] > 0.f || newmax) ? 1.f : 0.f;
}

__global__ void k_apply_mask(const float* maskf, const float* heat, float* out, int n)
{
    int i = blockIdx.x * blockDim.x + threadIdx.x;
    if (i < n) out[i] = (maskf[i] > 0.f) ? heat[i] : 0.f;
}

__global__ void k_zero_cnt(int* c) { if (threadIdx.x < 4) c[threadIdx.x] = 0; }

__global__ void collect_cand(const float* __restrict__ nms, float* __restrict__ cscore,
                             int* __restrict__ cidx, int* __restrict__ ccnt, int H, int W)
{
    int b = blockIdx.z;
    int i = blockIdx.x * blockDim.x + threadIdx.x;
    if (i >= H * W) return;
    int y = i / W, x = i - y * W;
    float v = nms[(size_t)b * H * W + i];
    bool inb = (y >= SP_BORDER && y < H - SP_BORDER && x >= SP_BORDER && x < W - SP_BORDER);
    if (inb && v > SP_THR) {
        int pos = atomicAdd(&ccnt[b], 1);
        if (pos < SP_CAP) {
            cscore[(size_t)b * SP_CAP + pos] = v;
            cidx[(size_t)b * SP_CAP + pos]   = i;
        }
    }
}

__global__ __launch_bounds__(1024)
void topk_select(float* __restrict__ cscore, const int* __restrict__ cidx,
                 const int* __restrict__ ccnt, float* __restrict__ vals,
                 float* __restrict__ kpts, int W)
{
    int b = blockIdx.x;
    int n = ccnt[b]; if (n > SP_CAP) n = SP_CAP;
    float* cs = cscore + (size_t)b * SP_CAP;
    const int* ci = cidx + (size_t)b * SP_CAP;

    __shared__ float sv[1024];
    __shared__ int   si[1024];
    __shared__ int   ssl[1024];

    for (int t = 0; t < SP_MAXK; ++t) {
        float bv = -1e30f; int bi = 0x7fffffff; int bs = -1;
        for (int j = threadIdx.x; j < n; j += 1024) {
            float v = cs[j]; int ix = ci[j];
            if (v > bv || (v == bv && ix < bi)) { bv = v; bi = ix; bs = j; }
        }
        sv[threadIdx.x] = bv; si[threadIdx.x] = bi; ssl[threadIdx.x] = bs;
        __syncthreads();
        for (int st = 512; st > 0; st >>= 1) {
            if (threadIdx.x < st) {
                float v = sv[threadIdx.x + st]; int ix = si[threadIdx.x + st];
                if (v > sv[threadIdx.x] || (v == sv[threadIdx.x] && ix < si[threadIdx.x])) {
                    sv[threadIdx.x] = v; si[threadIdx.x] = ix;
                    ssl[threadIdx.x] = ssl[threadIdx.x + st];
                }
            }
            __syncthreads();
        }
        if (threadIdx.x == 0) {
            float v = sv[0]; int ix = si[0]; int slot = ssl[0];
            if (slot < 0 || v <= 0.f) { v = 0.f; ix = 0; }
            else cs[slot] = -1e30f;
            vals[(size_t)b * SP_MAXK + t] = v;
            kpts[((size_t)b * SP_MAXK + t) * 2 + 0] = (float)(ix % W);
            kpts[((size_t)b * SP_MAXK + t) * 2 + 1] = (float)(ix / W);
        }
        __syncthreads();
    }
}

// d is NHWC: [B][P][C]
__global__ void desc_norm(const float* __restrict__ d, float* __restrict__ invn, int C, int P)
{
    int b = blockIdx.z;
    int i = blockIdx.x * blockDim.x + threadIdx.x;
    if (i >= P) return;
    const float* dp = d + ((size_t)b * P + i) * C;
    float s = 0.f;
    for (int c = 0; c < C; ++c) { float v = dp[c]; s += v * v; }
    invn[(size_t)b * P + i] = 1.f / fmaxf(sqrtf(s), 1e-12f);
}

__global__ __launch_bounds__(256)
void sample_desc(const float* __restrict__ d, const float* __restrict__ invn,
                 const float* __restrict__ kpts, float* __restrict__ out,
                 int C, int Hc, int Wc)
{
    int bk = blockIdx.x;
    int b = bk / SP_MAXK, k = bk - b * SP_MAXK;
    int c = threadIdx.x;

    float kx = kpts[(size_t)bk * 2 + 0];
    float ky = kpts[(size_t)bk * 2 + 1];
    const float sc = 8.f;
    float divx = Wc * sc - sc * 0.5f - 0.5f;
    float divy = Hc * sc - sc * 0.5f - 0.5f;
    float gx = (kx - sc * 0.5f + 0.5f) / divx * 2.f - 1.f;
    float gy = (ky - sc * 0.5f + 0.5f) / divy * 2.f - 1.f;
    float ix = (gx + 1.f) * 0.5f * (float)(Wc - 1);
    float iy = (gy + 1.f) * 0.5f * (float)(Hc - 1);
    float x0f = fminf(fmaxf(floorf(ix), 0.f), (float)(Wc - 1));
    float x1f = fminf(x0f + 1.f, (float)(Wc - 1));
    float y0f = fminf(fmaxf(floorf(iy), 0.f), (float)(Hc - 1));
    float y1f = fminf(y0f + 1.f, (float)(Hc - 1));
    float wx = ix - x0f, wy = iy - y0f;
    int x0 = (int)x0f, x1 = (int)x1f, y0 = (int)y0f, y1 = (int)y1f;

    int P = Hc * Wc;
    const float* dp = d + (size_t)b * P * C;
    const float* nb = invn + (size_t)b * P;
    int p00 = y0 * Wc + x0, p01 = y0 * Wc + x1;
    int p10 = y1 * Wc + x0, p11 = y1 * Wc + x1;
    float v00 = dp[(size_t)p00 * C + c] * nb[p00];
    float v01 = dp[(size_t)p01 * C + c] * nb[p01];
    float v10 = dp[(size_t)p10 * C + c] * nb[p10];
    float v11 = dp[(size_t)p11 * C + c] * nb[p11];
    float top = v00 * (1.f - wx) + v01 * wx;
    float bot = v10 * (1.f - wx) + v11 * wx;
    float v = top * (1.f - wy) + bot * wy;

    __shared__ float red[256];
    red[c] = v * v;
    __syncthreads();
    for (int st = 128; st > 0; st >>= 1) {
        if (c < st) red[c] += red[c + st];
        __syncthreads();
    }
    float nrm = fmaxf(sqrtf(red[0]), 1e-12f);
    out[((size_t)b * C + c) * SP_MAXK + k] = v / nrm;
}

// ---------------- host orchestration ----------------
extern "C" void kernel_launch(void* const* d_in, const int* in_sizes, int n_in,
                              void* d_out, int out_size, void* d_ws, size_t ws_size,
                              hipStream_t stream)
{
    (void)in_sizes; (void)n_in; (void)out_size; (void)ws_size;

    const int Bn = 4, IH = 480, IW = 640;
    const int HWf = IH * IW;               // 307200
    const int Hc = 60, Wc = 80, Pc = Hc * Wc;
    const int NHEAT = Bn * HWf;

    static const int LCIN[12]  = {1, 64, 64, 64, 64, 128, 128, 128, 128, 256, 128, 256};
    static const int LCOUT[12] = {64, 64, 64, 64, 128, 128, 128, 128, 256, 65, 256, 256};
    static const int LK[12]    = {3, 3, 3, 3, 3, 3, 3, 3, 3, 1, 3, 1};
    size_t woff[12], wtot = 0;
    for (int i = 0; i < 12; ++i) {
        woff[i] = wtot;
        size_t cnt = (size_t)LCOUT[i] * LCIN[i] * LK[i] * LK[i];
        wtot += (cnt + 63) & ~(size_t)63;          // keep layer bases 128B aligned
    }

    char* ws = (char*)d_ws;
    size_t off = 0;
    auto alloc = [&](size_t bytes) -> void* {
        void* p = ws + off;
        off += (bytes + 255) & ~(size_t)255;
        return p;
    };

    _Float16* w16   = (_Float16*)alloc(wtot * sizeof(_Float16));
    _Float16* img16 = (_Float16*)alloc((size_t)Bn * HWf * sizeof(_Float16));
    size_t actElems = (size_t)64 * HWf * Bn;          // biggest activation tensor
    _Float16* actA  = (_Float16*)alloc(actElems * sizeof(_Float16));
    _Float16* actB  = (_Float16*)alloc(actElems * sizeof(_Float16));
    float* semi  = (float*)alloc((size_t)Bn * 65 * Pc * sizeof(float));
    float* descf = (float*)alloc((size_t)Bn * 256 * Pc * sizeof(float));
    float* heat  = (float*)alloc((size_t)NHEAT * sizeof(float));
    float* tmpf  = (float*)alloc((size_t)NHEAT * sizeof(float));
    float* pm    = (float*)alloc((size_t)NHEAT * sizeof(float));
    float* ssb   = (float*)alloc((size_t)NHEAT * sizeof(float));
    float* ps    = (float*)alloc((size_t)NHEAT * sizeof(float));
    float* maskf = (float*)alloc((size_t)NHEAT * sizeof(float));
    float* nmsb  = (float*)alloc((size_t)NHEAT * sizeof(float));
    float* cscore = (float*)alloc((size_t)Bn * SP_CAP * sizeof(float));
    int*   cidx   = (int*)alloc((size_t)Bn * SP_CAP * sizeof(int));
    int*   ccnt   = (int*)alloc(64);
    float* invn   = (float*)alloc((size_t)Bn * Pc * sizeof(float));

    float* vals_out = (float*)d_out;                 // 4*1024
    float* kpts_out = vals_out + Bn * SP_MAXK;       // 4*1024*2
    float* desc_out = kpts_out + Bn * SP_MAXK * 2;   // 4*256*1024

    // 1) convert+reorder weights, convert image to f16 (image is NHWC with C=1)
    for (int i = 0; i < 12; ++i) {
        int cnt = LCOUT[i] * LCIN[i] * LK[i] * LK[i];
        convert_weights<<<(cnt + 255) / 256, 256, 0, stream>>>(
            (const float*)d_in[1 + 2 * i], w16 + woff[i], LCIN[i], LCOUT[i], LK[i]);
    }
    {
        int cnt = Bn * HWf;
        f32_to_f16<<<(cnt + 255) / 256, 256, 0, stream>>>(
            (const float*)d_in[0], img16, cnt);
    }

    auto ctz = [](int v) { int s = 0; while ((v & 1) == 0 && v > 1) { v >>= 1; ++s; } return s; };
    auto conv = [&](const _Float16* in, int li, _Float16* o16, float* o32,
                    int H, int W, int relu) {
        int Cin = LCIN[li], Cout = LCOUT[li], ks = LK[li];
        int pad = (ks == 3) ? 1 : 0;
        dim3 grid((H * W + TM - 1) / TM, (Cout + TN - 1) / TN, Bn);
        conv_wmma<<<grid, 256, 0, stream>>>(in, w16 + woff[li],
                                            (const float*)d_in[2 + 2 * li],
                                            o16, o32, Cin, ctz(Cin), Cout, H, W, ks, pad, relu);
    };
    auto pool = [&](const _Float16* in, _Float16* out, int C, int H, int W) {
        int n = C * (H / 2) * (W / 2);
        dim3 grid((n + 255) / 256, 1, Bn);
        maxpool2_f16<<<grid, 256, 0, stream>>>(in, out, C, ctz(C), H, W);
    };

    // 2) backbone (all activations NHWC)
    conv(img16, 0, actA, nullptr, 480, 640, 1);        // conv1a
    conv(actA,  1, actB, nullptr, 480, 640, 1);        // conv1b
    pool(actB, actA, 64, 480, 640);                    // -> 240x320
    conv(actA,  2, actB, nullptr, 240, 320, 1);        // conv2a
    conv(actB,  3, actA, nullptr, 240, 320, 1);        // conv2b
    pool(actA, actB, 64, 240, 320);                    // -> 120x160
    conv(actB,  4, actA, nullptr, 120, 160, 1);        // conv3a (64->128)
    conv(actA,  5, actB, nullptr, 120, 160, 1);        // conv3b
    pool(actB, actA, 128, 120, 160);                   // -> 60x80
    conv(actA,  6, actB, nullptr, 60, 80, 1);          // conv4a
    conv(actB,  7, actA, nullptr, 60, 80, 1);          // conv4b -> x lives in actA

    // 3) score head: convPa + convPb + softmax/pixel-shuffle
    conv(actA, 8, actB, nullptr, 60, 80, 1);           // convPa (128->256)
    conv(actB, 9, nullptr, semi, 60, 80, 0);           // convPb (256->65), f32 NHWC out
    {
        dim3 grid((Pc + 255) / 256, 1, Bn);
        softmax_heat<<<grid, 256, 0, stream>>>(semi, heat, Hc, Wc);
    }

    // 4) simple_nms (separable 9x9 max filters)
    dim3 gh((HWf + 255) / 256, 1, Bn);
    dim3 ge((NHEAT + 255) / 256);
    maxfilt_x<<<gh, 256, 0, stream>>>(heat, tmpf, IH, IW, SP_NMSR);
    maxfilt_y<<<gh, 256, 0, stream>>>(tmpf, pm,  IH, IW, SP_NMSR);
    k_eq_mask<<<ge, 256, 0, stream>>>(heat, pm, maskf, NHEAT);
    for (int it = 0; it < 2; ++it) {
        maxfilt_x<<<gh, 256, 0, stream>>>(maskf, tmpf, IH, IW, SP_NMSR);
        maxfilt_y<<<gh, 256, 0, stream>>>(tmpf, pm,   IH, IW, SP_NMSR);
        k_suppscore<<<ge, 256, 0, stream>>>(pm, heat, ssb, NHEAT);
        maxfilt_x<<<gh, 256, 0, stream>>>(ssb, tmpf, IH, IW, SP_NMSR);
        maxfilt_y<<<gh, 256, 0, stream>>>(tmpf, ps,  IH, IW, SP_NMSR);
        k_update_mask<<<ge, 256, 0, stream>>>(maskf, ssb, ps, pm, NHEAT);
    }
    k_apply_mask<<<ge, 256, 0, stream>>>(maskf, heat, nmsb, NHEAT);

    // 5) border/threshold + candidate compaction + top-k
    k_zero_cnt<<<1, 32, 0, stream>>>(ccnt);
    collect_cand<<<gh, 256, 0, stream>>>(nmsb, cscore, cidx, ccnt, IH, IW);
    topk_select<<<Bn, 1024, 0, stream>>>(cscore, cidx, ccnt, vals_out, kpts_out, IW);

    // 6) descriptor head: convDa + convDb + normalize + bilinear sample
    conv(actA, 10, actB, nullptr, 60, 80, 1);          // convDa (128->256)
    conv(actB, 11, nullptr, descf, 60, 80, 0);         // convDb (256->256), f32 NHWC out
    {
        dim3 grid((Pc + 255) / 256, 1, Bn);
        desc_norm<<<grid, 256, 0, stream>>>(descf, invn, 256, Pc);
    }
    sample_desc<<<Bn * SP_MAXK, 256, 0, stream>>>(descf, invn, kpts_out, desc_out,
                                                  256, Hc, Wc);
}
